// GlmImageAttention_78159814852906
// MI455X (gfx1250) — matrix operand order repaired and checked by reading, not compiled, not run on hardware
//
#include <hip/hip_runtime.h>

#define S_TOT   2304
#define S_TXT   256
#define DIM     3072
#define NHEAD   24
#define HDIM    128
#define INNER   3072
#define QKV_N   9216

typedef __attribute__((ext_vector_type(4)))  float  v4f;
typedef __attribute__((ext_vector_type(8)))  float  v8f;
typedef __attribute__((ext_vector_type(8)))  __bf16 v8bf;
typedef __attribute__((ext_vector_type(16))) __bf16 v16bf;

static __device__ __forceinline__ v8f v8f_zero() {
  v8f r;
  #pragma unroll
  for (int i = 0; i < 8; ++i) r[i] = 0.0f;
  return r;
}

static __device__ __forceinline__ v16bf bf16cat(v8bf lo, v8bf hi) {
  return __builtin_shufflevector(lo, hi, 0,1,2,3,4,5,6,7,8,9,10,11,12,13,14,15);
}

// A-fragment (16x32 bf16, row-major source):
// lane row = lane&15 ; grp0 holds K 0..7 & 16..23, grp1 holds K 8..15 & 24..31
static __device__ __forceinline__ v16bf ld_fragA(const __bf16* row, int grp) {
  v8bf lo = *(const v8bf*)(row + grp * 8);
  v8bf hi = *(const v8bf*)(row + grp * 8 + 16);
  return bf16cat(lo, hi);
}

// B-fragment (32x16 bf16) sourced from Bt[n][k] rows (k contiguous):
// lane col = lane&15 ; grp0 holds K 0..7 & 16..23, grp1 holds K 8..15 & 24..31
static __device__ __forceinline__ v16bf ld_fragB(const __bf16* row, int grp) {
  v8bf lo = *(const v8bf*)(row + grp * 8);
  v8bf hi = *(const v8bf*)(row + grp * 8 + 16);
  return bf16cat(lo, hi);
}

#define WMMA_BF16(a, b, c) \
  __builtin_amdgcn_wmma_f32_16x16x32_bf16(false, (a), false, (b), (short)0, (c), false, false)

// ---- gfx1250 async global->LDS copy (per-lane 16B), tracked by ASYNCcnt ----
// INST_OFFSET is added to both the LDS (VDST) and global (VADDR) addresses,
// so consecutive chunks share one address pair and use offset immediates.
static __device__ __forceinline__ void async_ld_32B(unsigned lds, const void* g) {
  asm volatile("global_load_async_to_lds_b128 %0, %1, off"           :: "v"(lds), "v"(g) : "memory");
  asm volatile("global_load_async_to_lds_b128 %0, %1, off offset:16" :: "v"(lds), "v"(g) : "memory");
}
static __device__ __forceinline__ void async_ld_64B(unsigned lds, const void* g) {
  asm volatile("global_load_async_to_lds_b128 %0, %1, off"           :: "v"(lds), "v"(g) : "memory");
  asm volatile("global_load_async_to_lds_b128 %0, %1, off offset:16" :: "v"(lds), "v"(g) : "memory");
  asm volatile("global_load_async_to_lds_b128 %0, %1, off offset:32" :: "v"(lds), "v"(g) : "memory");
  asm volatile("global_load_async_to_lds_b128 %0, %1, off offset:48" :: "v"(lds), "v"(g) : "memory");
}
#define ASYNC_WAIT(imm) asm volatile("s_wait_asynccnt " imm ::: "memory")
#define LDS_ADDR(p) ((unsigned)(size_t)(const void*)(p))

// ---------------------------------------------------------------------------
// Kernel 0a: concat(encoder, hidden) -> bf16 x buffer [2304][3072]
// ---------------------------------------------------------------------------
__global__ __launch_bounds__(256) void cvt_x_kernel(
    const float* __restrict__ enc, const float* __restrict__ hid,
    __bf16* __restrict__ xb)
{
  const int c    = blockIdx.x * 256 + threadIdx.x;  // 16-element chunk id
  const int base = c * 16;
  const int row  = base / DIM, col = base - row * DIM;
  const float* src = (row < S_TXT ? enc + (size_t)row * DIM
                                  : hid + (size_t)(row - S_TXT) * DIM) + col;
  __bf16* dst = xb + (size_t)row * DIM + col;
  #pragma unroll
  for (int j = 0; j < 16; j += 4) {
    v4f x = *(const v4f*)(src + j);
    dst[j+0] = (__bf16)x.x; dst[j+1] = (__bf16)x.y;
    dst[j+2] = (__bf16)x.z; dst[j+3] = (__bf16)x.w;
  }
}

// ---------------------------------------------------------------------------
// Kernel 0b: W[K][N] fp32 -> Wt[N][K] bf16 (64x64 LDS tile transpose)
// ---------------------------------------------------------------------------
__global__ __launch_bounds__(256) void wt_cvt_kernel(
    const float* __restrict__ W, __bf16* __restrict__ Wt, int Kdim, int Ndim)
{
  __shared__ __bf16 Ts[64 * 72];   // [n][k], stride 72
  const int t  = threadIdx.x;
  const int k0 = blockIdx.x * 64;
  const int n0 = blockIdx.y * 64;
  {
    const int kr = t >> 2, nc = (t & 3) * 16;
    const float* src = W + (size_t)(k0 + kr) * Ndim + n0 + nc;
    #pragma unroll
    for (int j = 0; j < 16; j += 4) {
      v4f x = *(const v4f*)(src + j);
      Ts[(nc + j + 0) * 72 + kr] = (__bf16)x.x;
      Ts[(nc + j + 1) * 72 + kr] = (__bf16)x.y;
      Ts[(nc + j + 2) * 72 + kr] = (__bf16)x.z;
      Ts[(nc + j + 3) * 72 + kr] = (__bf16)x.w;
    }
  }
  __syncthreads();
  {
    const int nr = t >> 2, kc = (t & 3) * 16;
    __bf16* dst = Wt + (size_t)(n0 + nr) * Kdim + k0 + kc;
    *(v8bf*)(dst)     = *(const v8bf*)(Ts + nr * 72 + kc);
    *(v8bf*)(dst + 8) = *(const v8bf*)(Ts + nr * 72 + kc + 8);
  }
}

// ---------------------------------------------------------------------------
// Kernel 1/4: C[M,N] = A[M,K](bf16) @ Bt[N,K](bf16)^T + bias, fp32 out.
// 128x128 block tile, K-step 32, double-buffered async global->LDS staging,
// last-iteration peeled (no in-loop branches around the async issue).
// 8 waves: 4 in M x 2 in N, wave tile 32x64 (8 wmma per K-step).
// ---------------------------------------------------------------------------
__global__ __launch_bounds__(256) void gemm_bf16_kernel(
    const __bf16* __restrict__ A, const __bf16* __restrict__ Bt,
    const float* __restrict__ bias, float* __restrict__ C,
    int N, int K, int permute)
{
  __shared__ __bf16 As[2][128 * 40];   // [m][k], row stride 40
  __shared__ __bf16 Bs[2][128 * 40];   // [n][k], row stride 40
  const int t    = threadIdx.x;
  const int lane = t & 31;
  const int wave = t >> 5;
  const int lr   = lane & 15;
  const int grp  = (lane >> 4) & 1;
  const int wm   = wave & 3;
  const int wn   = wave >> 2;
  const int mBase = blockIdx.y * 128;
  const int nBase = blockIdx.x * 128;

  v8f acc[2][4];
  #pragma unroll
  for (int i = 0; i < 2; ++i)
    #pragma unroll
    for (int j = 0; j < 4; ++j) acc[i][j] = v8f_zero();

  // staging map: thread t copies 32B of one 64B (32 bf16) tile row
  const int srow = t >> 1, soff = (t & 1) * 16;   // element offset 0 or 16
  const __bf16* aP = A  + (size_t)(mBase + srow) * K + soff;
  const __bf16* bP = Bt + (size_t)(nBase + srow) * K + soff;
  const unsigned lA0 = LDS_ADDR(&As[0][srow * 40 + soff]);
  const unsigned lA1 = LDS_ADDR(&As[1][srow * 40 + soff]);
  const unsigned lB0 = LDS_ADDR(&Bs[0][srow * 40 + soff]);
  const unsigned lB1 = LDS_ADDR(&Bs[1][srow * 40 + soff]);

  auto compute = [&](int buf) {
    v16bf af0 = ld_fragA(&As[buf][(wm * 32 + lr) * 40], grp);
    v16bf af1 = ld_fragA(&As[buf][(wm * 32 + 16 + lr) * 40], grp);
    #pragma unroll
    for (int nt = 0; nt < 4; ++nt) {
      v16bf bf = ld_fragB(&Bs[buf][(wn * 64 + nt * 16 + lr) * 40], grp);
      acc[0][nt] = WMMA_BF16(af0, bf, acc[0][nt]);
      acc[1][nt] = WMMA_BF16(af1, bf, acc[1][nt]);
    }
  };

  const int KT = K / 32;
  // prologue: stage tile 0 into buffer 0
  async_ld_32B(lA0, aP);
  async_ld_32B(lB0, bP);
  aP += 32; bP += 32;

  for (int kt = 0; kt < KT - 1; ++kt) {
    const int buf = kt & 1;
    __syncthreads();                       // all waves done reading buf^1
    async_ld_32B(buf ? lA0 : lA1, aP);     // prefetch tile kt+1
    async_ld_32B(buf ? lB0 : lB1, bP);
    aP += 32; bP += 32;
    ASYNC_WAIT("0x4");                     // tile kt landed (kt+1 in flight)
    __syncthreads();                       // tile kt visible block-wide
    compute(buf);
  }
  ASYNC_WAIT("0x0");                       // last tile landed
  __syncthreads();
  compute((KT - 1) & 1);

  // epilogue: bias + store (C layout: M = v + 8*grp, N = lane&15).
  // 128-row tiles never straddle the 256-row text/image split -> uniform adj.
  const int rowAdj = permute ? ((mBase < S_TXT) ? (S_TOT - S_TXT) : -S_TXT) : 0;
  #pragma unroll
  for (int nt = 0; nt < 4; ++nt) {
    const int n  = nBase + wn * 64 + nt * 16 + lr;
    const float bv = bias[n];
    #pragma unroll
    for (int mt = 0; mt < 2; ++mt) {
      const int m0 = mBase + wm * 32 + mt * 16 + 8 * grp + rowAdj;
      #pragma unroll
      for (int v = 0; v < 8; ++v)
        C[(size_t)(m0 + v) * N + n] = acc[mt][nt][v] + bv;
    }
  }
}

// ---------------------------------------------------------------------------
// Kernel 2: per-(token,head) layernorm(q,k) + RoPE(image tokens) + bf16 pack.
// q scaled by 1/sqrt(128). v stored transposed [h][d][s]. One wave per (s,h).
// ---------------------------------------------------------------------------
__global__ __launch_bounds__(256) void ln_rope_pack_kernel(
    const float* __restrict__ qkv, const float* __restrict__ cosb,
    const float* __restrict__ sinb, __bf16* __restrict__ qb,
    __bf16* __restrict__ kb, __bf16* __restrict__ vbT)
{
  const int t    = threadIdx.x;
  const int lane = t & 31;
  const int wave = t >> 5;
  const int p    = blockIdx.x * 8 + wave;
  const int s    = p / NHEAD;
  const int h    = p - s * NHEAD;
  const int d0   = lane * 4;
  const float* row = qkv + (size_t)s * QKV_N + h * HDIM;

  #pragma unroll
  for (int which = 0; which < 2; ++which) {   // 0 = q, 1 = k
    v4f x = *(const v4f*)(row + which * INNER + d0);
    float sum = x.x + x.y + x.z + x.w;
    #pragma unroll
    for (int off = 16; off > 0; off >>= 1) sum += __shfl_xor(sum, off, 32);
    float mu = sum * (1.0f / 128.0f);
    float a0 = x.x - mu, a1 = x.y - mu, a2 = x.z - mu, a3 = x.w - mu;
    float vs = a0*a0 + a1*a1 + a2*a2 + a3*a3;
    #pragma unroll
    for (int off = 16; off > 0; off >>= 1) vs += __shfl_xor(vs, off, 32);
    float rs = rsqrtf(vs * (1.0f / 128.0f) + 1e-5f);
    float y0 = a0 * rs, y1 = a1 * rs, y2 = a2 * rs, y3 = a3 * rs;
    if (which == 0) {
      const float qs = 0.08838834764831845f;  // 1/sqrt(HEAD_DIM)
      y0 *= qs; y1 *= qs; y2 *= qs; y3 *= qs;
    }
    if (s >= S_TXT) {                          // interleaved RoPE
      int pos = s - S_TXT;
      int i0  = d0 >> 1;
      float c0 = cosb[pos * 64 + i0],     sn0 = sinb[pos * 64 + i0];
      float c1 = cosb[pos * 64 + i0 + 1], sn1 = sinb[pos * 64 + i0 + 1];
      float o0 = y0 * c0 - y1 * sn0;
      float o1 = y1 * c0 + y0 * sn0;
      float o2 = y2 * c1 - y3 * sn1;
      float o3 = y3 * c1 + y2 * sn1;
      y0 = o0; y1 = o1; y2 = o2; y3 = o3;
    }
    __bf16* dst = (which == 0 ? qb : kb) + ((size_t)h * S_TOT + s) * HDIM + d0;
    dst[0] = (__bf16)y0; dst[1] = (__bf16)y1;
    dst[2] = (__bf16)y2; dst[3] = (__bf16)y3;
  }
  // v: plain bf16 convert, transposed store [h][d][s]
  v4f xv = *(const v4f*)(row + 2 * INNER + d0);
  size_t vb = ((size_t)h * HDIM + d0) * S_TOT + s;
  vbT[vb]             = (__bf16)xv.x;
  vbT[vb + S_TOT]     = (__bf16)xv.y;
  vbT[vb + 2 * S_TOT] = (__bf16)xv.z;
  vbT[vb + 3 * S_TOT] = (__bf16)xv.w;
}

// ---------------------------------------------------------------------------
// Kernel 3: flash attention. One head per blockIdx.y, 64 queries per block
// (4 waves x 16 rows). 32-key K/V tiles double-buffered via async->LDS,
// last iteration peeled. Online softmax in the WMMA C layout.
// ---------------------------------------------------------------------------
__global__ __launch_bounds__(128) void attention_kernel(
    const __bf16* __restrict__ qb, const __bf16* __restrict__ kb,
    const __bf16* __restrict__ vbT, __bf16* __restrict__ attnb)
{
  __shared__ __bf16 Ks[2][32 * 136];   // [key][d], stride 136
  __shared__ __bf16 Vt[2][128 * 40];   // [d][key], stride 40
  __shared__ __bf16 Pw[4 * 16 * 40];   // per-wave P tile [q][key], stride 40
  const int t    = threadIdx.x;
  const int lane = t & 31;
  const int wave = t >> 5;
  const int lr   = lane & 15;
  const int grp  = (lane >> 4) & 1;
  const int h    = blockIdx.y;
  const int q0   = blockIdx.x * 64 + wave * 16;

  // q A-fragments for the 4 d-blocks of K=128 (loaded once from global)
  v16bf qf[4];
  const __bf16* qrow = qb + ((size_t)h * S_TOT + q0 + lr) * HDIM;
  #pragma unroll
  for (int db = 0; db < 4; ++db) qf[db] = ld_fragA(qrow + db * 32, grp);

  v8f od[8];
  #pragma unroll
  for (int j = 0; j < 8; ++j) od[j] = v8f_zero();
  float mrow[8], lsum[8];
  #pragma unroll
  for (int v = 0; v < 8; ++v) { mrow[v] = -1e30f; lsum[v] = 0.0f; }

  // staging map (128 threads, 8 async b128 per thread per tile):
  const int krow = t >> 2, kq = (t & 3) * 32;   // K: row, 64B quarter
  const __bf16* kP = kb  + ((size_t)h * S_TOT + krow) * HDIM + kq;
  const __bf16* vP = vbT + ((size_t)h * HDIM + t) * S_TOT;
  const unsigned lK0 = LDS_ADDR(&Ks[0][krow * 136 + kq]);
  const unsigned lK1 = LDS_ADDR(&Ks[1][krow * 136 + kq]);
  const unsigned lV0 = LDS_ADDR(&Vt[0][t * 40]);
  const unsigned lV1 = LDS_ADDR(&Vt[1][t * 40]);

  auto process_tile = [&](int buf) {
    // scores: 16 queries x 32 keys (2 n-tiles x 4 d-steps)
    v8f sacc[2];
    sacc[0] = v8f_zero(); sacc[1] = v8f_zero();
    #pragma unroll
    for (int nt = 0; nt < 2; ++nt) {
      const __bf16* krowp = &Ks[buf][(nt * 16 + lr) * 136];
      #pragma unroll
      for (int db = 0; db < 4; ++db) {
        v16bf bf = ld_fragB(krowp + db * 32, grp);
        sacc[nt] = WMMA_BF16(qf[db], bf, sacc[nt]);
      }
    }
    // online softmax per query row (16-lane-group reductions)
    float alpha[8];
    #pragma unroll
    for (int v = 0; v < 8; ++v) {
      float tm = fmaxf(sacc[0][v], sacc[1][v]);
      tm = fmaxf(tm, __shfl_xor(tm, 1, 32));
      tm = fmaxf(tm, __shfl_xor(tm, 2, 32));
      tm = fmaxf(tm, __shfl_xor(tm, 4, 32));
      tm = fmaxf(tm, __shfl_xor(tm, 8, 32));
      float nm = fmaxf(mrow[v], tm);
      float al = __expf(mrow[v] - nm);
      float p0 = __expf(sacc[0][v] - nm);
      float p1 = __expf(sacc[1][v] - nm);
      sacc[0][v] = p0; sacc[1][v] = p1;
      float ps = p0 + p1;
      ps += __shfl_xor(ps, 1, 32);
      ps += __shfl_xor(ps, 2, 32);
      ps += __shfl_xor(ps, 4, 32);
      ps += __shfl_xor(ps, 8, 32);
      lsum[v] = lsum[v] * al + ps;
      mrow[v] = nm;
      alpha[v] = al;
    }
    #pragma unroll
    for (int j = 0; j < 8; ++j)
      #pragma unroll
      for (int v = 0; v < 8; ++v) od[j][v] *= alpha[v];
    // transpose P (C layout -> row-major [q][key]) via wave-private LDS
    __bf16* P = Pw + wave * 16 * 40;
    #pragma unroll
    for (int nt = 0; nt < 2; ++nt)
      #pragma unroll
      for (int v = 0; v < 8; ++v)
        P[(v + 8 * grp) * 40 + nt * 16 + lr] = (__bf16)sacc[nt][v];
    // out += P @ V : one A-frag (K=32 keys), 8 d-column tiles
    v16bf pf = ld_fragA(P + lr * 40, grp);
    #pragma unroll
    for (int j = 0; j < 8; ++j) {
      v16bf vf = ld_fragB(&Vt[buf][(j * 16 + lr) * 40], grp);
      od[j] = WMMA_BF16(pf, vf, od[j]);
    }
  };

  const int NT = S_TOT / 32;   // 72 key tiles
  // prologue: stage tile 0
  async_ld_64B(lK0, kP);
  async_ld_64B(lV0, vP);
  kP += 32 * HDIM; vP += 32;

  for (int it = 0; it < NT - 1; ++it) {
    const int buf = it & 1;
    __syncthreads();                       // all waves done reading buf^1
    async_ld_64B(buf ? lK0 : lK1, kP);     // prefetch tile it+1
    async_ld_64B(buf ? lV0 : lV1, vP);
    kP += 32 * HDIM; vP += 32;
    ASYNC_WAIT("0x8");                     // tile it landed
    __syncthreads();
    process_tile(buf);
  }
  ASYNC_WAIT("0x0");
  __syncthreads();
  process_tile((NT - 1) & 1);

  // finalize: divide by row sums, store bf16 [s][h*128 + d]
  #pragma unroll
  for (int j = 0; j < 8; ++j) {
    const int n = h * HDIM + j * 16 + lr;
    #pragma unroll
    for (int v = 0; v < 8; ++v) {
      int q = q0 + v + 8 * grp;
      attnb[(size_t)q * INNER + n] = (__bf16)(od[j][v] / lsum[v]);
    }
  }
}

// ---------------------------------------------------------------------------
extern "C" void kernel_launch(void* const* d_in, const int* in_sizes, int n_in,
                              void* d_out, int out_size, void* d_ws, size_t ws_size,
                              hipStream_t stream) {
  (void)in_sizes; (void)n_in; (void)out_size; (void)ws_size;
  const float* hidden  = (const float*)d_in[0];  // [1,2048,3072]
  const float* encoder = (const float*)d_in[1];  // [1,256,3072]
  const float* cosb    = (const float*)d_in[2];  // [2048,64]
  const float* sinb    = (const float*)d_in[3];  // [2048,64]
  const float* Wqkv    = (const float*)d_in[4];  // [3072,9216]
  const float* bqkv    = (const float*)d_in[5];  // [9216]
  const float* Wout    = (const float*)d_in[6];  // [3072,3072]
  const float* bout    = (const float*)d_in[7];  // [3072]
  // d_in[8] = text_seq_length (256, hardcoded)

  char* w = (char*)d_ws;
  size_t off = 0;
  float*  qkvbuf = (float*)(w + off);                  // 2304*9216 fp32
  __bf16* attnb  = (__bf16*)(w + off);                 // reuse (qkv dead by then)
  off += (size_t)S_TOT * QKV_N * sizeof(float);
  __bf16* qbb = (__bf16*)(w + off); off += (size_t)NHEAD * S_TOT * HDIM * 2;
  __bf16* kbb = (__bf16*)(w + off); off += (size_t)NHEAD * S_TOT * HDIM * 2;
  __bf16* vbT = (__bf16*)(w + off); off += (size_t)NHEAD * S_TOT * HDIM * 2;
  __bf16* xb    = (__bf16*)(w + off); off += (size_t)S_TOT * DIM * 2;
  __bf16* WqkvT = (__bf16*)(w + off); off += (size_t)QKV_N * DIM * 2;
  __bf16* WoutT = (__bf16*)(w + off); off += (size_t)INNER * DIM * 2;

  // 0: one-time bf16 conversion / weight transposition (memory-bound, ~10us)
  cvt_x_kernel<<<dim3(S_TOT * DIM / 16 / 256), 256, 0, stream>>>(encoder, hidden, xb);
  wt_cvt_kernel<<<dim3(DIM / 64, QKV_N / 64), 256, 0, stream>>>(Wqkv, WqkvT, DIM, QKV_N);
  wt_cvt_kernel<<<dim3(INNER / 64, DIM / 64), 256, 0, stream>>>(Wout, WoutT, INNER, DIM);

  // 1: qkv = x @ Wqkv + b
  gemm_bf16_kernel<<<dim3(QKV_N / 128, S_TOT / 128), 256, 0, stream>>>(
      xb, WqkvT, bqkv, qkvbuf, QKV_N, DIM, 0);

  // 2: layernorm + rope + pack
  ln_rope_pack_kernel<<<dim3(S_TOT * NHEAD / 8), 256, 0, stream>>>(
      qkvbuf, cosb, sinb, qbb, kbb, vbT);

  // 3: flash attention -> bf16
  attention_kernel<<<dim3(S_TOT / 64, NHEAD), 128, 0, stream>>>(
      qbb, kbb, vbT, attnb);

  // 4: out = attn @ Wout + b, rows permuted into (hidden_out, encoder_out)
  gemm_bf16_kernel<<<dim3(DIM / 128, S_TOT / 128), 256, 0, stream>>>(
      attnb, WoutT, bout, (float*)d_out, DIM, INNER, 1);
}
